// Simple_graph_net_49709951484366
// MI455X (gfx1250) — compile-verified, matrix operand order checked by reading, not run on hardware
//
#include <hip/hip_runtime.h>

typedef __attribute__((ext_vector_type(16))) _Float16 v16h;
typedef __attribute__((ext_vector_type(8)))  float    v8f;

#define GDIM 64
#define CELL 16
#define IMW  1024
#define TSZ  32
#define CH   40
#define KTOT 360   // 9*40 im2col depth for conv2
#define KPAD 384   // 12 k-steps of 32
#define NPAD 48    // 3 n-tiles of 16
#define A1W  17    // act1 padded width (SAME pad_hi = 1 row/col of zeros)
#define ZIDX ((16 * A1W + 16) * CH)   // zeroed pad cell, 16B-aligned redirect target

__device__ __forceinline__ float gelu_tanh(float x) {
    float x3 = x * x * x;
    return 0.5f * x * (1.0f + tanhf(0.7978845608028654f * (x + 0.044715f * x3)));
}

union FragU { v16h v; uint4 q[2]; };

// Kernel 0: conv2_w [ky][kx][ci][co] (f32) -> Bt[n=co][k=(ky,kx,ci)] f16, zero padded.
// conv2_w flat index is exactly k*CH + n.
__global__ void prep_b_kernel(const float* __restrict__ w2, _Float16* __restrict__ Bt) {
    int idx = blockIdx.x * blockDim.x + threadIdx.x;
    if (idx >= NPAD * KPAD) return;
    int n = idx / KPAD;
    int k = idx - n * KPAD;
    float v = 0.0f;
    if (n < CH && k < KTOT) v = w2[k * CH + n];
    Bt[idx] = (_Float16)v;
}

__global__ __launch_bounds__(128)
void fused_net_kernel(const float* __restrict__ img,
                      const float* __restrict__ msk,
                      const float* __restrict__ w1,
                      const float* __restrict__ b1,
                      const float* __restrict__ s1,
                      const float* __restrict__ bb1,
                      const float* __restrict__ b2,
                      const float* __restrict__ s2,
                      const float* __restrict__ bb2,
                      const float* __restrict__ dw,
                      const float* __restrict__ db,
                      const _Float16* __restrict__ Bt,
                      float* __restrict__ out) {
    __shared__ float tok[33 * 33];                          // padded 32x32 token
    __shared__ float w1s[KTOT];                             // conv1 weights [tap][c]
    __shared__ float prm[240];                              // b1,s1,bb1,b2,s2,bb2
    __shared__ __align__(16) _Float16 act1[A1W * A1W * CH]; // conv1 out, physically padded
    __shared__ float y2[64 * NPAD];                         // conv2 pre-activation
    __shared__ float act2[64 * CH];                         // conv2 activations (flatten order)
    __shared__ float pp[128 * 6];                           // dense partials

    const int tid = threadIdx.x;
    const int f   = blockIdx.x;              // global token id
    const int cfg = f >> 10;                 // config order: (0,0),(1,0),(0,1),(1,1)
    const int t   = f & 1023;
    const int ti  = t >> 5;
    const int tj  = t & 31;
    const int sx  = cfg & 1;
    const int sy  = cfg >> 1;
    const float cid = (float)((2 * ti + sx) * GDIM + (2 * tj + sy));

    // ---- phase 1: masked shifted token -> LDS; weights/params -> LDS; zero act1 pad ----
    for (int e = tid; e < 33 * 33; e += 128) {
        int u = e / 33, v = e - (e / 33) * 33;
        float val = 0.0f;
        if (u < 32 && v < 32) {
            int oy = (TSZ * ti + u + CELL * sx) & (IMW - 1);
            int ox = (TSZ * tj + v + CELL * sy) & (IMW - 1);
            float m = msk[oy * IMW + ox];
            val = (m == cid) ? img[oy * IMW + ox] : 0.0f;
        }
        tok[e] = val;
    }
    for (int e = tid; e < KTOT; e += 128) w1s[e] = w1[e];
    // zero the 33 padded act1 cells (row 16 full + col 16 partial): 33*CH halves
    for (int e = tid; e < 33 * CH; e += 128) {
        int pos = e / CH, c = e - (e / CH) * CH;
        int yy, xx;
        if (pos < A1W) { yy = 16; xx = pos; } else { yy = pos - A1W; xx = 16; }
        act1[(yy * A1W + xx) * CH + c] = (_Float16)0.0f;
    }
    if (tid < CH) {
        prm[tid]       = b1[tid];
        prm[40 + tid]  = s1[tid];
        prm[80 + tid]  = bb1[tid];
        prm[120 + tid] = b2[tid];
        prm[160 + tid] = s2[tid];
        prm[200 + tid] = bb2[tid];
    }
    __syncthreads();

    // ---- phase 2: conv1 (3x3 s2 SAME, 1->40) + LN + GELU -> act1 (f16) ----
    for (int rep = 0; rep < 2; ++rep) {
        int p = tid + rep * 128;
        int py = p >> 4, px = p & 15;
        int iy = 2 * py, ix = 2 * px;
        float t00 = tok[(iy+0)*33 + ix+0], t01 = tok[(iy+0)*33 + ix+1], t02 = tok[(iy+0)*33 + ix+2];
        float t10 = tok[(iy+1)*33 + ix+0], t11 = tok[(iy+1)*33 + ix+1], t12 = tok[(iy+1)*33 + ix+2];
        float t20 = tok[(iy+2)*33 + ix+0], t21 = tok[(iy+2)*33 + ix+1], t22 = tok[(iy+2)*33 + ix+2];
        float yv[CH];
        float sum = 0.f, sq = 0.f;
        #pragma unroll
        for (int c = 0; c < CH; ++c) {
            float y = prm[c];
            y += t00 * w1s[0*CH+c] + t01 * w1s[1*CH+c] + t02 * w1s[2*CH+c];
            y += t10 * w1s[3*CH+c] + t11 * w1s[4*CH+c] + t12 * w1s[5*CH+c];
            y += t20 * w1s[6*CH+c] + t21 * w1s[7*CH+c] + t22 * w1s[8*CH+c];
            yv[c] = y; sum += y; sq += y * y;
        }
        float mu  = sum * (1.0f / CH);
        float var = sq  * (1.0f / CH) - mu * mu;
        float rs  = rsqrtf(var + 1e-6f);
        #pragma unroll
        for (int c = 0; c < CH; ++c) {
            float h = (yv[c] - mu) * rs * prm[40 + c] + prm[80 + c];
            act1[(py * A1W + px) * CH + c] = (_Float16)gelu_tanh(h);
        }
    }
    __syncthreads();

    // ---- phase 3: conv2 as GEMM via WMMA: [64 x 360] x [360 x 40] ----
    const int lane  = tid & 31;
    const int mt    = tid >> 5;              // wave = m-tile
    const int m     = mt * 16 + (lane & 15);
    const int py2   = m >> 3, px2 = m & 7;
    const int khalf = (lane >> 4) * 8;       // ISA 16-bit A layout: upper lanes hold K+8
    const int nrow  = lane & 15;

    v8f acc0 = {}, acc1 = {}, acc2 = {};

    // branch-free A group load: k guaranteed < KTOT, padding handled by act1 layout
    auto loadA = [&](int k) -> uint4 {
        int tap = k / CH;
        int ci  = k - tap * CH;
        int ky  = tap / 3, kx = tap - ky * 3;
        int yy  = 2 * py2 + ky, xx = 2 * px2 + kx;   // <= 16, lands in zero pad when 16
        return *(const uint4*)&act1[(yy * A1W + xx) * CH + ci];
    };

    #pragma unroll
    for (int ks = 0; ks < 11; ++ks) {        // K = 0..351: all groups valid
        int kb = ks * 32 + khalf;
        FragU af;
        af.q[0] = loadA(kb);
        af.q[1] = loadA(kb + 16);
        FragU bf0, bf1, bf2;
        bf0.q[0] = *(const uint4*)&Bt[( 0 + nrow) * KPAD + kb];
        bf0.q[1] = *(const uint4*)&Bt[( 0 + nrow) * KPAD + kb + 16];
        bf1.q[0] = *(const uint4*)&Bt[(16 + nrow) * KPAD + kb];
        bf1.q[1] = *(const uint4*)&Bt[(16 + nrow) * KPAD + kb + 16];
        bf2.q[0] = *(const uint4*)&Bt[(32 + nrow) * KPAD + kb];
        bf2.q[1] = *(const uint4*)&Bt[(32 + nrow) * KPAD + kb + 16];
        acc0 = __builtin_amdgcn_wmma_f32_16x16x32_f16(false, af.v, false, bf0.v, (short)0, acc0, false, false);
        acc1 = __builtin_amdgcn_wmma_f32_16x16x32_f16(false, af.v, false, bf1.v, (short)0, acc1, false, false);
        acc2 = __builtin_amdgcn_wmma_f32_16x16x32_f16(false, af.v, false, bf2.v, (short)0, acc2, false, false);
    }

    {   // ks = 11: K = 352..383. Lower group valid only for lanes 0-15 (K=352..359,
        // tap=8 -> ky=kx=2, ci=32); upper group (K=368..383) is zero for all lanes.
        int kb = 352 + khalf;
        int yy = 2 * py2 + 2, xx = 2 * px2 + 2;
        int idx = (yy * A1W + xx) * CH + 32;
        idx = (khalf == 0) ? idx : ZIDX;     // v_cndmask on address, EXEC stays full
        FragU af;
        af.q[0] = *(const uint4*)&act1[idx];
        af.q[1] = make_uint4(0u, 0u, 0u, 0u);
        FragU bf0, bf1, bf2;                 // Bt is zero-padded beyond K=360
        bf0.q[0] = *(const uint4*)&Bt[( 0 + nrow) * KPAD + kb];
        bf0.q[1] = *(const uint4*)&Bt[( 0 + nrow) * KPAD + kb + 16];
        bf1.q[0] = *(const uint4*)&Bt[(16 + nrow) * KPAD + kb];
        bf1.q[1] = *(const uint4*)&Bt[(16 + nrow) * KPAD + kb + 16];
        bf2.q[0] = *(const uint4*)&Bt[(32 + nrow) * KPAD + kb];
        bf2.q[1] = *(const uint4*)&Bt[(32 + nrow) * KPAD + kb + 16];
        acc0 = __builtin_amdgcn_wmma_f32_16x16x32_f16(false, af.v, false, bf0.v, (short)0, acc0, false, false);
        acc1 = __builtin_amdgcn_wmma_f32_16x16x32_f16(false, af.v, false, bf1.v, (short)0, acc1, false, false);
        acc2 = __builtin_amdgcn_wmma_f32_16x16x32_f16(false, af.v, false, bf2.v, (short)0, acc2, false, false);
    }

    {   // dump C (VGPR v -> row v / v+8, lane%16 -> col)
        int rbase = mt * 16 + ((lane >> 4) << 3);
        int col   = lane & 15;
        #pragma unroll
        for (int v = 0; v < 8; ++v) {
            y2[(rbase + v) * NPAD + col]      = acc0[v];
            y2[(rbase + v) * NPAD + 16 + col] = acc1[v];
            y2[(rbase + v) * NPAD + 32 + col] = acc2[v];
        }
    }
    __syncthreads();

    // ---- phase 4: +bias, LN, GELU -> act2 (f32, flatten order [pos][ch]) ----
    if (tid < 64) {
        float sum = 0.f, sq = 0.f;
        #pragma unroll
        for (int c = 0; c < CH; ++c) {
            float z = y2[tid * NPAD + c] + prm[120 + c];
            sum += z; sq += z * z;
        }
        float mu  = sum * (1.0f / CH);
        float var = sq  * (1.0f / CH) - mu * mu;
        float rs  = rsqrtf(var + 1e-6f);
        #pragma unroll
        for (int c = 0; c < CH; ++c) {
            float z = y2[tid * NPAD + c] + prm[120 + c];
            float h = (z - mu) * rs * prm[160 + c] + prm[200 + c];
            act2[tid * CH + c] = gelu_tanh(h);
        }
    }
    __syncthreads();

    // ---- phase 5: dense 2560 -> 6 ----
    {
        float part[6] = {0.f, 0.f, 0.f, 0.f, 0.f, 0.f};
        int kb = tid * 20;                    // 128 threads x 20 = 2560
        for (int kk = 0; kk < 20; ++kk) {
            int k = kb + kk;
            float a = act2[k];
            const float* wr = &dw[k * 6];
            #pragma unroll
            for (int co = 0; co < 6; ++co) part[co] += a * wr[co];
        }
        #pragma unroll
        for (int co = 0; co < 6; ++co) pp[tid * 6 + co] = part[co];
    }
    __syncthreads();
    if (tid < 6) {
        float s = db[tid];
        for (int tt = 0; tt < 128; ++tt) s += pp[tt * 6 + tid];
        out[f * 6 + tid] = s;
    }
}

extern "C" void kernel_launch(void* const* d_in, const int* in_sizes, int n_in,
                              void* d_out, int out_size, void* d_ws, size_t ws_size,
                              hipStream_t stream) {
    const float* img = (const float*)d_in[0];
    const float* msk = (const float*)d_in[1];
    const float* w1  = (const float*)d_in[2];
    const float* b1  = (const float*)d_in[3];
    const float* s1  = (const float*)d_in[4];
    const float* bb1 = (const float*)d_in[5];
    const float* w2  = (const float*)d_in[6];
    const float* b2  = (const float*)d_in[7];
    const float* s2  = (const float*)d_in[8];
    const float* bb2 = (const float*)d_in[9];
    const float* dw  = (const float*)d_in[10];
    const float* db  = (const float*)d_in[11];
    float* out = (float*)d_out;
    _Float16* Bt = (_Float16*)d_ws;           // 48*384*2 = 36864 bytes of scratch

    prep_b_kernel<<<(NPAD * KPAD + 255) / 256, 256, 0, stream>>>(w2, Bt);
    fused_net_kernel<<<4096, 128, 0, stream>>>(img, msk, w1, b1, s1, bb1,
                                               b2, s2, bb2, dw, db, Bt, out);
}
